// DynamicConvDecoder_18253611008113
// MI455X (gfx1250) — compile-verified, matrix operand order checked by reading, not run on hardware
//
#include <hip/hip_runtime.h>
#include <math.h>

// ---------------------------------------------------------------------------
// Model constants (from reference)
// ---------------------------------------------------------------------------
#define Cdim   1024
#define Hh     16
#define DHd    64
#define FFd    4096
#define Vv     32000
#define Tt     512
#define Bb     4
#define Mrow   (Tt * Bb)          // 2048 token rows, layout (t*B + b)

typedef __attribute__((ext_vector_type(16))) __bf16        bf16x16;
typedef __attribute__((ext_vector_type(8)))  float         f32x8;
typedef __attribute__((ext_vector_type(8)))  unsigned int  u32x8;
typedef __attribute__((ext_vector_type(4)))  unsigned int  u32x4;
typedef __attribute__((ext_vector_type(8)))  int           i32x8;
typedef __attribute__((ext_vector_type(4)))  int           i32x4;

#if defined(__has_builtin)
#if __has_builtin(__builtin_amdgcn_tensor_load_to_lds) && \
    __has_builtin(__builtin_amdgcn_s_wait_tensorcnt)
#define USE_TDM 1
#endif
#endif

__device__ __forceinline__ unsigned short f32_to_bf16(float f) {
  unsigned int u = __builtin_bit_cast(unsigned int, f);
  unsigned int r = (u + 0x7FFFu + ((u >> 16) & 1u)) >> 16;
  return (unsigned short)r;
}

// ---------------------------------------------------------------------------
// Direct-global WMMA GEMM (fallback for ragged shapes):
// Out[M,N] (+)= alpha * (A[M,K](bf16) @ Bt[N,K]^T(bf16) + bias[N])
// flags: bit0 = ReLU, bit1 = accumulate into Out
// Fragment layout (CDNA5 ISA, 16-bit A 16x32): lane L holds row m=L&15
// (A) / col n=L&15 (B); VGPR v holds K pair k0,k0+1 with
//   k0 = (v>>2)*16 + (L>>4)*8 + (v&3)*2
// C/D layout: VGPR r holds rows r (lanes 0-15) and r+8 (lanes 16-31), n=L&15.
// ---------------------------------------------------------------------------
__global__ __launch_bounds__(256)
void gemm_bf16_kernel(const unsigned short* __restrict__ A,
                      const unsigned short* __restrict__ Bt,
                      const float* __restrict__ bias,
                      float* __restrict__ Out,
                      int M, int N, int K, int ldOut, float alpha, int flags) {
  const int lane = threadIdx.x & 31;
  const int wave = threadIdx.x >> 5;
  const int mBase = blockIdx.y * 64  + (wave >> 2) * 32;  // 2 waves in M
  const int nBase = blockIdx.x * 128 + (wave & 3) * 32;   // 4 waves in N
  const int lm = lane & 15;
  const int lk = lane >> 4;

  f32x8 acc[2][2] = {};

  int rowA[2], rowB[2];
#pragma unroll
  for (int i = 0; i < 2; ++i) {
    int r = mBase + i * 16 + lm; rowA[i] = (r < M) ? r : (M - 1);  // clamped rows are
    int c = nBase + i * 16 + lm; rowB[i] = (c < N) ? c : (N - 1);  // never stored
  }

  for (int kb = 0; kb < K; kb += 32) {
    u32x8 af[2], bfr[2];
#pragma unroll
    for (int i = 0; i < 2; ++i) {
      const unsigned int* ap = (const unsigned int*)(A  + (size_t)rowA[i] * K + kb);
      const unsigned int* bp = (const unsigned int*)(Bt + (size_t)rowB[i] * K + kb);
      if (kb + 32 < K) {                      // lowers to global_prefetch_b8
        __builtin_prefetch(ap + 16, 0, 1);
        __builtin_prefetch(bp + 16, 0, 1);
      }
#pragma unroll
      for (int v = 0; v < 8; ++v) {
        const int k0 = ((v >> 2) << 4) + (lk << 3) + ((v & 3) << 1);
        af[i][v]  = ap[k0 >> 1];
        bfr[i][v] = bp[k0 >> 1];
      }
    }
#pragma unroll
    for (int i = 0; i < 2; ++i) {
      bf16x16 av = __builtin_bit_cast(bf16x16, af[i]);
#pragma unroll
      for (int j = 0; j < 2; ++j) {
        bf16x16 bv = __builtin_bit_cast(bf16x16, bfr[j]);
        acc[i][j] = __builtin_amdgcn_wmma_f32_16x16x32_bf16(
            false, av, false, bv, (short)0, acc[i][j], false, false);
      }
    }
  }

#pragma unroll
  for (int j = 0; j < 2; ++j) {
    const int n = nBase + j * 16 + lm;
    if (n >= N) continue;
    const float bv = bias ? bias[n] : 0.0f;
#pragma unroll
    for (int i = 0; i < 2; ++i) {
#pragma unroll
      for (int r = 0; r < 8; ++r) {
        const int m = mBase + i * 16 + lk * 8 + r;
        if (m < M) {
          float val = alpha * (acc[i][j][r] + bv);
          if (flags & 1) val = fmaxf(val, 0.0f);
          const size_t o = (size_t)m * ldOut + n;
          if (flags & 2) Out[o] += val; else Out[o] = val;
        }
      }
    }
  }
}

#ifdef USE_TDM
// ---------------------------------------------------------------------------
// TDM descriptor issue: DMA a (128 rows x 32 cols) bf16 tile, row stride K
// elements, from global gptr into LDS at byte offset ldsOff.
// D# packing per CDNA5 ISA (group0) / §8.4 (group1):
//   g0: [1:0]=count=1, [63:32]=lds_addr, [120:64]=global_addr, [127:126]=type 2
//   g1: [17:16]=data_size=1 (2B), [79:48]=tensor_dim0=K, [111:80]=tensor_dim1,
//       [127:112]=tile_dim0=32, [143:128]=tile_dim1=128, [207:160]=dim0_stride=K
// groups 2/3 zero (2D tensor).  clang-23 6-arg builtin form:
//   (u32x4 g0, i32x8 g1, i32x4, i32x4, i32x8, i32 cpol)
// ---------------------------------------------------------------------------
__device__ __forceinline__ void tdm_issue(const unsigned short* gptr, unsigned ldsOff,
                                          int K, int rows) {
  const unsigned long long ga = (unsigned long long)(size_t)gptr;
  u32x4 g0;
  g0[0] = 1u;                                              // count=1, user mode
  g0[1] = ldsOff;                                          // lds_addr (bytes)
  g0[2] = (unsigned)(ga & 0xFFFFFFFFull);                  // global_addr[31:0]
  g0[3] = (unsigned)((ga >> 32) & 0x1FFFFFFull) | 0x80000000u;  // [56:32] | type=2
  i32x8 g1;
  g1[0] = 1 << 16;                                         // data_size = 2 bytes
  g1[1] = (K & 0xFFFF) << 16;                              // tensor_dim0 lo16
  g1[2] = ((K >> 16) & 0xFFFF) | ((rows & 0xFFFF) << 16);  // dim0 hi / dim1 lo
  g1[3] = ((rows >> 16) & 0xFFFF) | (32 << 16);            // dim1 hi / tile_dim0=32
  g1[4] = 128;                                             // tile_dim1=128, dim2=0
  g1[5] = K;                                               // tensor_dim0_stride lo32
  g1[6] = 0;                                               // stride hi / dim1_stride
  g1[7] = 0;
  const i32x4 z4 = {0, 0, 0, 0};
  const i32x8 z8 = {0, 0, 0, 0, 0, 0, 0, 0};
  __builtin_amdgcn_tensor_load_to_lds(g0, g1, z4, z4, z8, 0);
}

// ---------------------------------------------------------------------------
// TDM-staged WMMA GEMM for aligned shapes (M%128==0, N%128==0, K%32==0).
// Block tile 128x128 (8 waves as 4(M) x 2(N); wave tile 32x64 = 2x4 WMMA).
// Double-buffered LDS tiles filled by the Tensor Data Mover (wave 0 issues,
// s_wait_tensorcnt + workgroup barriers order produce/consume/reuse).
// Fragments come from LDS as two ds_load_b128 per fragment.
// ---------------------------------------------------------------------------
__global__ __launch_bounds__(256)
void gemm_bf16_tdm_kernel(const unsigned short* __restrict__ A,
                          const unsigned short* __restrict__ Bt,
                          const float* __restrict__ bias,
                          float* __restrict__ Out,
                          int M, int N, int K, int ldOut, float alpha, int flags) {
  __shared__ unsigned short Abuf[2][128 * 32];   // 8 KiB each
  __shared__ unsigned short Bbuf[2][128 * 32];
  const int lane  = threadIdx.x & 31;
  const int wave  = threadIdx.x >> 5;
  const int waveM = wave & 3;                    // 4 waves in M
  const int waveN = wave >> 2;                   // 2 waves in N
  const int lm = lane & 15;
  const int lk = lane >> 4;
  const int mBlk = blockIdx.y * 128;
  const int nBlk = blockIdx.x * 128;

  f32x8 acc[2][4] = {};

  if (wave == 0) {
    tdm_issue(A  + (size_t)mBlk * K, (unsigned)(size_t)&Abuf[0][0], K, M);
    tdm_issue(Bt + (size_t)nBlk * K, (unsigned)(size_t)&Bbuf[0][0], K, N);
  }

  for (int kb = 0; kb < K; kb += 32) {
    const int buf = (kb >> 5) & 1;
    if (wave == 0) __builtin_amdgcn_s_wait_tensorcnt(0);
    __syncthreads();                             // tiles visible to all waves
    if (wave == 0 && kb + 32 < K) {              // prefetch next K panel into buf^1
      tdm_issue(A  + (size_t)mBlk * K + (kb + 32),
                (unsigned)(size_t)&Abuf[buf ^ 1][0], K, M);
      tdm_issue(Bt + (size_t)nBlk * K + (kb + 32),
                (unsigned)(size_t)&Bbuf[buf ^ 1][0], K, N);
    }

    u32x8 af[2], bfr[4];
#pragma unroll
    for (int i = 0; i < 2; ++i) {
      const unsigned short* ap = &Abuf[buf][(waveM * 32 + i * 16 + lm) * 32];
      const u32x4 lo = *(const u32x4*)(ap + lk * 8);
      const u32x4 hi = *(const u32x4*)(ap + lk * 8 + 16);
      af[i] = __builtin_shufflevector(lo, hi, 0, 1, 2, 3, 4, 5, 6, 7);
    }
#pragma unroll
    for (int j = 0; j < 4; ++j) {
      const unsigned short* bp = &Bbuf[buf][(waveN * 64 + j * 16 + lm) * 32];
      const u32x4 lo = *(const u32x4*)(bp + lk * 8);
      const u32x4 hi = *(const u32x4*)(bp + lk * 8 + 16);
      bfr[j] = __builtin_shufflevector(lo, hi, 0, 1, 2, 3, 4, 5, 6, 7);
    }
#pragma unroll
    for (int i = 0; i < 2; ++i) {
      bf16x16 av = __builtin_bit_cast(bf16x16, af[i]);
#pragma unroll
      for (int j = 0; j < 4; ++j) {
        bf16x16 bv = __builtin_bit_cast(bf16x16, bfr[j]);
        acc[i][j] = __builtin_amdgcn_wmma_f32_16x16x32_bf16(
            false, av, false, bv, (short)0, acc[i][j], false, false);
      }
    }
    __syncthreads();                             // reads done before buf reuse
  }

#pragma unroll
  for (int j = 0; j < 4; ++j) {
    const int n = nBlk + waveN * 64 + j * 16 + lm;
    if (n >= N) continue;
    const float bv = bias ? bias[n] : 0.0f;
#pragma unroll
    for (int i = 0; i < 2; ++i) {
#pragma unroll
      for (int r = 0; r < 8; ++r) {
        const int m = mBlk + waveM * 32 + i * 16 + lk * 8 + r;
        if (m < M) {
          float val = alpha * (acc[i][j][r] + bv);
          if (flags & 1) val = fmaxf(val, 0.0f);
          const size_t o = (size_t)m * ldOut + n;
          if (flags & 2) Out[o] += val; else Out[o] = val;
        }
      }
    }
  }
}
#endif  // USE_TDM

// ---------------------------------------------------------------------------
// fp32 -> bf16 elementwise
// ---------------------------------------------------------------------------
__global__ __launch_bounds__(256)
void tobf16_kernel(const float* __restrict__ src, unsigned short* __restrict__ dst,
                   size_t n) {
  size_t i = (size_t)blockIdx.x * blockDim.x + threadIdx.x;
  if (i < n) dst[i] = f32_to_bf16(src[i]);
}

// ---------------------------------------------------------------------------
// fp32 (R,Ccols) -> bf16 transposed (Ccols,R)   (weight pre-transpose)
// ---------------------------------------------------------------------------
__global__ __launch_bounds__(256)
void transpose_f32_bf16_kernel(const float* __restrict__ src,
                               unsigned short* __restrict__ dst,
                               int R, int Ccols) {
  __shared__ float tile[32][33];
  const int tx = threadIdx.x, ty = threadIdx.y;     // 32 x 8
  const int c0 = blockIdx.x * 32, r0 = blockIdx.y * 32;
  for (int i = 0; i < 32; i += 8) {
    const int r = r0 + ty + i, c = c0 + tx;
    tile[ty + i][tx] = (r < R && c < Ccols) ? src[(size_t)r * Ccols + c] : 0.0f;
  }
  __syncthreads();
  for (int i = 0; i < 32; i += 8) {
    const int c = c0 + ty + i, r = r0 + tx;
    if (c < Ccols && r < R) dst[(size_t)c * R + r] = f32_to_bf16(tile[tx][ty + i]);
  }
}

// ---------------------------------------------------------------------------
// Embedding gather + LayerNorm -> x fp32 (rows = t*B+b).  blockDim=256, C=1024
// ---------------------------------------------------------------------------
__global__ __launch_bounds__(256)
void embed_ln_kernel(const int* __restrict__ prev, const float* __restrict__ we,
                     const float* __restrict__ pe, const float* __restrict__ te,
                     const float* __restrict__ g, const float* __restrict__ b,
                     float* __restrict__ x) {
  __shared__ float red[256];
  const int row = blockIdx.x;                  // t*B + bb
  const int t = row / Bb, bb = row % Bb;
  const int tok = prev[bb * Tt + t];
  const int tid = threadIdx.x;
  float vals[4], s = 0.f, s2 = 0.f;
#pragma unroll
  for (int i = 0; i < 4; ++i) {
    const int c = tid + i * 256;
    const float v = we[(size_t)tok * Cdim + c] + pe[(size_t)t * Cdim + c] + te[c];
    vals[i] = v; s += v; s2 += v * v;
  }
  red[tid] = s; __syncthreads();
  for (int st = 128; st > 0; st >>= 1) { if (tid < st) red[tid] += red[tid + st]; __syncthreads(); }
  const float mean = red[0] / Cdim; __syncthreads();
  red[tid] = s2; __syncthreads();
  for (int st = 128; st > 0; st >>= 1) { if (tid < st) red[tid] += red[tid + st]; __syncthreads(); }
  const float var = red[0] / Cdim - mean * mean;
  const float inv = rsqrtf(var + 1e-5f);
#pragma unroll
  for (int i = 0; i < 4; ++i) {
    const int c = tid + i * 256;
    x[(size_t)row * Cdim + c] = (vals[i] - mean) * inv * g[c] + b[c];
  }
}

// ---------------------------------------------------------------------------
// LayerNorm fp32 in -> bf16 out.  permute!=0 remaps row t*B+b -> b*T+t (final)
// ---------------------------------------------------------------------------
__global__ __launch_bounds__(256)
void layernorm_kernel(const float* __restrict__ in, const float* __restrict__ g,
                      const float* __restrict__ b, unsigned short* __restrict__ outb,
                      int permute) {
  __shared__ float red[256];
  const int row = blockIdx.x;
  const int tid = threadIdx.x;
  const float* xr = in + (size_t)row * Cdim;
  float s = 0.f, s2 = 0.f;
#pragma unroll
  for (int i = 0; i < 4; ++i) {
    const float v = xr[tid + i * 256]; s += v; s2 += v * v;
  }
  red[tid] = s; __syncthreads();
  for (int st = 128; st > 0; st >>= 1) { if (tid < st) red[tid] += red[tid + st]; __syncthreads(); }
  const float mean = red[0] / Cdim; __syncthreads();
  red[tid] = s2; __syncthreads();
  for (int st = 128; st > 0; st >>= 1) { if (tid < st) red[tid] += red[tid + st]; __syncthreads(); }
  const float var = red[0] / Cdim - mean * mean;
  const float inv = rsqrtf(var + 1e-5f);
  int orow = row;
  if (permute) { const int t = row / Bb, bb = row % Bb; orow = bb * Tt + t; }
#pragma unroll
  for (int i = 0; i < 4; ++i) {
    const int c = tid + i * 256;
    outb[(size_t)orow * Cdim + c] = f32_to_bf16((xr[c] - mean) * inv * g[c] + b[c]);
  }
}

// ---------------------------------------------------------------------------
// GLU: in (M,2C) -> a*sigmoid(g): out fp32 + bf16 (M,C)
// ---------------------------------------------------------------------------
__global__ __launch_bounds__(256)
void glu_kernel(const float* __restrict__ in, float* __restrict__ outf,
                unsigned short* __restrict__ outb, size_t n) {
  size_t i = (size_t)blockIdx.x * blockDim.x + threadIdx.x;
  if (i >= n) return;
  const size_t row = i / Cdim, c = i % Cdim;
  const float a  = in[row * (2 * Cdim) + c];
  const float gg = in[row * (2 * Cdim) + Cdim + c];
  const float v = a * (1.0f / (1.0f + __expf(-gg)));
  outf[i] = v; outb[i] = f32_to_bf16(v);
}

// ---------------------------------------------------------------------------
// xc[:, j*C:(j+1)*C] = x  for j=0..3  (residual pre-fill of concat buffer)
// ---------------------------------------------------------------------------
__global__ __launch_bounds__(256)
void bcast4_kernel(const float* __restrict__ x, float* __restrict__ xc, size_t n) {
  size_t i = (size_t)blockIdx.x * blockDim.x + threadIdx.x;
  if (i >= n) return;
  const size_t row = i / (4 * Cdim), c = i % Cdim;
  xc[i] = x[row * Cdim + c];
}

// ---------------------------------------------------------------------------
// DynamicConv: per-(t,b,h) softmax over K taps (LDS), causal sum over taps.
// y   : GLU output fp32 (M,C), w : y@Wdyn fp32 (M, H*K), out fp32 (M,C)
// ---------------------------------------------------------------------------
__global__ __launch_bounds__(256)
void dynconv_kernel(const float* __restrict__ y, const float* __restrict__ w,
                    float* __restrict__ out, int Ktaps) {
  __shared__ float wt[Hh * 31];
  const int row = blockIdx.x;                 // t*B + bb
  const int t = row / Bb, bb = row % Bb;
  const int tid = threadIdx.x;
  if (tid < Hh) {
    const float* wr = w + (size_t)row * (Hh * Ktaps) + tid * Ktaps;
    float mx = -1e30f;
    for (int k = 0; k < Ktaps; ++k) mx = fmaxf(mx, wr[k]);
    float sum = 0.f;
    for (int k = 0; k < Ktaps; ++k) { const float e = __expf(wr[k] - mx); wt[tid * Ktaps + k] = e; sum += e; }
    const float inv = 1.0f / sum;
    for (int k = 0; k < Ktaps; ++k) wt[tid * Ktaps + k] *= inv;
  }
  __syncthreads();
  for (int c = tid; c < Cdim; c += 256) {
    const int h = c / DHd;
    float acc = 0.f;
    for (int k = 0; k < Ktaps; ++k) {
      const int ts = t - (Ktaps - 1) + k;
      if (ts >= 0) acc += wt[h * Ktaps + k] * y[((size_t)ts * Bb + bb) * Cdim + c];
    }
    out[(size_t)row * Cdim + c] = acc;
  }
}

// ---------------------------------------------------------------------------
// Cross-attention core: one block per (t,b,h).  q pre-scaled by DH^-0.5.
// q rows t*B+b (cols h*64..), k/v rows s*B+b.  mask (B,S) bool, True = pad.
// ---------------------------------------------------------------------------
__global__ __launch_bounds__(256)
void attention_kernel(const float* __restrict__ q, const float* __restrict__ k,
                      const float* __restrict__ v, const unsigned char* __restrict__ mask,
                      float* __restrict__ o, int S) {
  __shared__ float sc[512];
  __shared__ float qv[DHd];
  __shared__ float red[256];
  const int idx = blockIdx.x;
  const int h  = idx % Hh;
  const int bb = (idx / Hh) % Bb;
  const int t  = idx / (Hh * Bb);
  const int tid = threadIdx.x;
  const size_t qoff = ((size_t)t * Bb + bb) * Cdim + h * DHd;
  if (tid < DHd) qv[tid] = q[qoff + tid];
  __syncthreads();
  for (int s = tid; s < S; s += 256) {
    const float* kr = k + ((size_t)s * Bb + bb) * Cdim + h * DHd;
    float d = 0.f;
#pragma unroll
    for (int e = 0; e < DHd; ++e) d += qv[e] * kr[e];
    if (mask[bb * S + s]) d = -1e9f;
    sc[s] = d;
  }
  __syncthreads();
  float lm = -1e30f;
  for (int s = tid; s < S; s += 256) lm = fmaxf(lm, sc[s]);
  red[tid] = lm; __syncthreads();
  for (int st = 128; st > 0; st >>= 1) { if (tid < st) red[tid] = fmaxf(red[tid], red[tid + st]); __syncthreads(); }
  const float mx = red[0]; __syncthreads();
  float ls = 0.f;
  for (int s = tid; s < S; s += 256) { const float e = __expf(sc[s] - mx); sc[s] = e; ls += e; }
  red[tid] = ls; __syncthreads();
  for (int st = 128; st > 0; st >>= 1) { if (tid < st) red[tid] += red[tid + st]; __syncthreads(); }
  const float inv = 1.0f / red[0];
  __syncthreads();
  if (tid < DHd) {
    float acc = 0.f;
    for (int s = 0; s < S; ++s) acc += sc[s] * v[((size_t)s * Bb + bb) * Cdim + h * DHd + tid];
    o[qoff + tid] = acc * inv;
  }
}

// ---------------------------------------------------------------------------
// Host orchestration
// ---------------------------------------------------------------------------
// Input flattening assumption: jax pytree order (dict keys sorted at every
// level).  Top level: ctx_article, ctx_faces, ctx_image, ctx_obj,
// mask_article, mask_faces, mask_image, mask_obj, <params subtree>, prev_target.
// params subtree (sorted): emb_ln_b, emb_ln_g, embed_out, final_ln_b,
// final_ln_g, layers[0..3], pos_emb, tok_emb, word_emb.
// Per-layer (55 leaves, sorted): W1, W2, Wc, Wdyn, Wf1, Wf2,
//   att_article{Wk,Wo,Wq,Wv,bk,bo,bq,bv}, att_faces{...}, att_image{...},
//   att_obj{...}, b1, b2, bc, bf1, bf2, ln_article_b, ln_article_g,
//   ln_conv_b, ln_conv_g, ln_f_b, ln_f_g, ln_faces_b, ln_faces_g,
//   ln_image_b, ln_image_g, ln_obj_b, ln_obj_g.
extern "C" void kernel_launch(void* const* d_in, const int* in_sizes, int n_in,
                              void* d_out, int out_size, void* d_ws, size_t ws_size,
                              hipStream_t stream) {
  (void)in_sizes; (void)n_in; (void)out_size; (void)ws_size;

  auto F = [&](int i) -> const float* { return (const float*)d_in[i]; };

  // ---- workspace layout (bytes) ----
  char* ws = (char*)d_ws;
  constexpr size_t SZ_ROW   = (size_t)Mrow * Cdim * 4;          // 8 MiB
  constexpr size_t OFF_X    = 0;
  constexpr size_t OFF_T0   = OFF_X  + SZ_ROW;
  constexpr size_t OFF_T1   = OFF_T0 + SZ_ROW;
  constexpr size_t OFF_T2   = OFF_T1 + SZ_ROW;
  constexpr size_t OFF_T3   = OFF_T2 + SZ_ROW;
  constexpr size_t OFF_BIG  = OFF_T3 + SZ_ROW;                  // M x 4C f32
  constexpr size_t OFF_ABF  = OFF_BIG + (size_t)Mrow * 4 * Cdim * 4;  // M x 4C bf16
  constexpr size_t OFF_WBF  = OFF_ABF + (size_t)Mrow * 4 * Cdim * 2;  // V x C bf16
  constexpr size_t OFF_CBF  = OFF_WBF + (size_t)Vv * Cdim * 2;
  constexpr size_t CTX_ELEMS = (size_t)49*Bb*2048 + (size_t)512*Bb*1024
                             + (size_t)16*Bb*512  + (size_t)36*Bb*2048;
  constexpr size_t OFF_Y2   = OFF_CBF + CTX_ELEMS * 2;

  float*          X    = (float*)(ws + OFF_X);
  float*          T0   = (float*)(ws + OFF_T0);
  float*          T1   = (float*)(ws + OFF_T1);
  float*          T2   = (float*)(ws + OFF_T2);
  float*          T3   = (float*)(ws + OFF_T3);
  float*          BIG  = (float*)(ws + OFF_BIG);
  unsigned short* ABF  = (unsigned short*)(ws + OFF_ABF);
  unsigned short* WBF  = (unsigned short*)(ws + OFF_WBF);
  unsigned short* CBF  = (unsigned short*)(ws + OFF_CBF);
  float*          Y2   = (float*)(ws + OFF_Y2);

  auto gemm = [&](const unsigned short* A, const unsigned short* Bt, const float* bias,
                  float* Out, int M, int N, int K, int ldOut, float alpha, int flags) {
#ifdef USE_TDM
    if ((M % 128) == 0 && (N % 128) == 0 && (K % 32) == 0) {
      dim3 g(N / 128, M / 128);
      gemm_bf16_tdm_kernel<<<g, 256, 0, stream>>>(A, Bt, bias, Out, M, N, K, ldOut, alpha, flags);
      return;
    }
#endif
    dim3 g((N + 127) / 128, (M + 63) / 64);
    gemm_bf16_kernel<<<g, 256, 0, stream>>>(A, Bt, bias, Out, M, N, K, ldOut, alpha, flags);
  };
  auto transposeW = [&](const float* W, int R, int Ccols) {  // -> WBF (Ccols,R) bf16
    dim3 b(32, 8), g((Ccols + 31) / 32, (R + 31) / 32);
    transpose_f32_bf16_kernel<<<g, b, 0, stream>>>(W, WBF, R, Ccols);
  };
  auto tobf16 = [&](const float* s, unsigned short* d, size_t n) {
    tobf16_kernel<<<(unsigned)((n + 255) / 256), 256, 0, stream>>>(s, d, n);
  };

  // ---- contexts (CTX order: image, article, faces, obj) ----
  struct CtxInfo { int ctxIdx, maskIdx, attOff, lngOff, lnbOff, kdim, S; size_t bfOff; };
  const CtxInfo ctxs[4] = {
    // ctxIdx maskIdx attOff ln_g ln_b  kdim   S    bf16 offset (elements)
    {2, 6, 22, 52, 51, 2048,  49, 0},
    {0, 4,  6, 44, 43, 1024, 512, (size_t)49*Bb*2048},
    {1, 5, 14, 50, 49,  512,  16, (size_t)49*Bb*2048 + (size_t)512*Bb*1024},
    {3, 7, 30, 54, 53, 2048,  36, (size_t)49*Bb*2048 + (size_t)512*Bb*1024 + (size_t)16*Bb*512},
  };
  static const int KS[4] = {3, 7, 15, 31};

  // convert contexts to bf16 once (reused by all 4 layers)
  for (int j = 0; j < 4; ++j)
    tobf16(F(ctxs[j].ctxIdx), CBF + ctxs[j].bfOff, (size_t)ctxs[j].S * Bb * ctxs[j].kdim);

  // ---- embedding + LN ----
  const int* prev = (const int*)d_in[236];
  embed_ln_kernel<<<Mrow, 256, 0, stream>>>(prev, F(235) /*word_emb*/, F(233) /*pos_emb*/,
                                            F(234) /*tok_emb[0]*/, F(9) /*emb_ln_g*/,
                                            F(8) /*emb_ln_b*/, X);

  // ---- layers ----
  for (int L = 0; L < 4; ++L) {
    const int lb = 13 + L * 55;
    const int Ktaps = KS[L];
    const float *W1 = F(lb+0), *W2 = F(lb+1), *Wc = F(lb+2), *Wdyn = F(lb+3);
    const float *Wf1 = F(lb+4), *Wf2 = F(lb+5);
    const float *b1 = F(lb+38), *b2 = F(lb+39), *bc = F(lb+40), *bf1 = F(lb+41), *bf2 = F(lb+42);
    const float *ln_conv_b = F(lb+45), *ln_conv_g = F(lb+46);
    const float *ln_f_b = F(lb+47), *ln_f_g = F(lb+48);

    // --- dynamic conv block ---
    layernorm_kernel<<<Mrow, 256, 0, stream>>>(X, ln_conv_g, ln_conv_b, ABF, 0);
    transposeW(W1, Cdim, 2 * Cdim);
    gemm(ABF, WBF, b1, BIG, Mrow, 2 * Cdim, Cdim, 2 * Cdim, 1.0f, 0);
    glu_kernel<<<(Mrow * Cdim) / 256, 256, 0, stream>>>(BIG, T0, ABF, (size_t)Mrow * Cdim);
    transposeW(Wdyn, Cdim, Hh * Ktaps);
    gemm(ABF, WBF, nullptr, Y2, Mrow, Hh * Ktaps, Cdim, Hh * Ktaps, 1.0f, 0);
    dynconv_kernel<<<Mrow, 256, 0, stream>>>(T0, Y2, T2, Ktaps);
    tobf16(T2, ABF, (size_t)Mrow * Cdim);
    transposeW(W2, Cdim, Cdim);
    gemm(ABF, WBF, b2, X, Mrow, Cdim, Cdim, Cdim, 1.0f, 2 /*x += res*/);

    // --- 4 cross-attention branches into concat buffer ---
    bcast4_kernel<<<(Mrow * 4 * Cdim) / 256, 256, 0, stream>>>(X, BIG, (size_t)Mrow * 4 * Cdim);
    for (int j = 0; j < 4; ++j) {
      const CtxInfo& ci = ctxs[j];
      const int ab = lb + ci.attOff;
      const float *Wk = F(ab+0), *Wo = F(ab+1), *Wq = F(ab+2), *Wv = F(ab+3);
      const float *bk = F(ab+4), *bo = F(ab+5), *bq = F(ab+6), *bv = F(ab+7);
      const int Mkv = ci.S * Bb;
      layernorm_kernel<<<Mrow, 256, 0, stream>>>(X, F(lb+ci.lngOff), F(lb+ci.lnbOff), ABF, 0);
      transposeW(Wq, Cdim, Cdim);
      gemm(ABF, WBF, bq, T1, Mrow, Cdim, Cdim, Cdim, 0.125f /*DH^-0.5*/, 0);
      transposeW(Wk, ci.kdim, Cdim);
      gemm(CBF + ci.bfOff, WBF, bk, T2, Mkv, Cdim, ci.kdim, Cdim, 1.0f, 0);
      transposeW(Wv, ci.kdim, Cdim);
      gemm(CBF + ci.bfOff, WBF, bv, T3, Mkv, Cdim, ci.kdim, Cdim, 1.0f, 0);
      attention_kernel<<<Bb * Hh * Tt, 256, 0, stream>>>(
          T1, T2, T3, (const unsigned char*)d_in[ci.maskIdx], T0, ci.S);
      tobf16(T0, ABF, (size_t)Mrow * Cdim);
      transposeW(Wo, Cdim, Cdim);
      gemm(ABF, WBF, bo, BIG + j * Cdim, Mrow, Cdim, Cdim, 4 * Cdim, 1.0f, 2 /*+= res*/);
    }
    // concat @ Wc + bc -> new x
    tobf16(BIG, ABF, (size_t)Mrow * 4 * Cdim);
    transposeW(Wc, 4 * Cdim, Cdim);
    gemm(ABF, WBF, bc, X, Mrow, Cdim, 4 * Cdim, Cdim, 1.0f, 0);

    // --- FFN ---
    layernorm_kernel<<<Mrow, 256, 0, stream>>>(X, ln_f_g, ln_f_b, ABF, 0);
    transposeW(Wf1, Cdim, FFd);
    gemm(ABF, WBF, bf1, BIG, Mrow, FFd, Cdim, FFd, 1.0f, 1 /*ReLU*/);
    tobf16(BIG, ABF, (size_t)Mrow * FFd);
    transposeW(Wf2, FFd, Cdim);
    gemm(ABF, WBF, bf2, X, Mrow, Cdim, FFd, Cdim, 1.0f, 2 /*x += res*/);
  }

  // ---- final LN (permute (t,b)->(b,t)) + output projection ----
  layernorm_kernel<<<Mrow, 256, 0, stream>>>(X, F(12) /*final_ln_g*/, F(11) /*final_ln_b*/, ABF, 1);
  // logits = x @ embed_out^T : embed_out (V,C) is already the (N,K) layout
  tobf16(F(10) /*embed_out*/, WBF, (size_t)Vv * Cdim);
  gemm(ABF, WBF, nullptr, (float*)d_out, Mrow, Vv, Cdim, Vv, 1.0f, 0);
}
